// GatedGNN_36687610642604
// MI455X (gfx1250) — matrix operand order repaired and checked by reading, not compiled, not run on hardware
//
#include <hip/hip_runtime.h>
#include <hip/hip_bf16.h>

typedef _Float16 v16h __attribute__((ext_vector_type(16)));
typedef float    v8f  __attribute__((ext_vector_type(8)));

#define GN_NODES 100000
#define GN_EDGES 1600000
#define GNFEAT   500
#define GNHID    128
#define GNCLASS  40
#define GNLAYERS 3

#define BM 64
#define BN 128
#define BK 32

union H2 { _Float16 h[2]; unsigned int u; };

// ---------------------------------------------------------------------------
// f16-WMMA GEMM: C[M,N] = A[M,K] @ op(B) + bias
//   A fp32 row-major [M,K];  opB==0: B fp32 [K,N];  opB==1: B fp32 [N,K]
// Block: 256 threads (8 waves), tile 64x128, K-step 32.
// Wave (wm 0..3, wn 0..1) owns a 16x64 strip: ONE A fragment reused by FOUR
// B fragments -> four v_wmma_f32_16x16x32_f16 per K-step (high A reuse;
// cuts A re-reads for the N=384 GRU GEMMs from 6x to 3x).
// Staging is branch-free (clamped addresses + cndmask-zero) and vectorized
// (global_load_b64 float2 -> packed 2xf16 ds_store_b32). Requires K even.
// ---------------------------------------------------------------------------
__global__ __launch_bounds__(256) void wmma_gemm_bias(
    const float* __restrict__ A, const float* __restrict__ B,
    const float* __restrict__ bias, float* __restrict__ C,
    int M, int N, int K, int opB)
{
    __shared__ _Float16 As[BM][BK];   // row-major A tile (f16)        4 KB
    __shared__ _Float16 Bs[BN][BK];   // B tile stored n-major (B^T)   8 KB

    const int tid  = threadIdx.x;
    const int wave = tid >> 5;        // 0..7
    const int lane = tid & 31;
    const int wm   = wave & 3;        // 4 waves along M (16 rows each)
    const int wn   = wave >> 2;       // 2 waves along N (64 cols each)

    const int bm = blockIdx.x * BM;
    const int bn = blockIdx.y * BN;

    const int g   = lane >> 4;        // lane group 0/1
    const int l16 = lane & 15;

    v8f acc[4] = {};

    // staging coordinates (fixed per thread)
    const int c2 = (tid & 15) * 2;    // even k offset within tile
    const int r0 = tid >> 4;          // 0..15

    for (int k0 = 0; k0 < K; k0 += BK) {
        // ---- stage A tile 64x32: 4 x (float2 load -> packed f16 store) ----
        {
            const int  gk  = k0 + c2;
            const bool kOK = gk < K;              // K even -> covers gk+1
            const long kIdx = kOK ? gk : 0;
            float2 v[4];
#pragma unroll
            for (int i = 0; i < 4; ++i) {
                const int  gm   = bm + r0 + 16 * i;
                const long mIdx = (gm < M) ? gm : (M - 1);
                v[i] = *(const float2*)(A + mIdx * K + kIdx);
            }
#pragma unroll
            for (int i = 0; i < 4; ++i) {
                const int  gm = bm + r0 + 16 * i;
                const bool ok = kOK && (gm < M);
                H2 p;
                p.h[0] = (_Float16)(ok ? v[i].x : 0.0f);
                p.h[1] = (_Float16)(ok ? v[i].y : 0.0f);
                *(unsigned int*)&As[r0 + 16 * i][c2] = p.u;
            }
        }
        // ---- stage B tile 128x32 -> Bs[n][k] (transposed) ----
        if (opB == 0) {
            // B[K,N] row-major: coalesced in n, scalar transpose stores
            const int n   = tid & 127;            // 0..127
            const int kk0 = tid >> 7;             // 0..1
            const int  gn   = bn + n;
            const long nIdx = (gn < N) ? gn : (N - 1);
            float v[16];
#pragma unroll
            for (int i = 0; i < 16; ++i) {
                const int  kk   = kk0 + 2 * i;
                const int  gk   = k0 + kk;
                const long kIdx = (gk < K) ? gk : 0;
                v[i] = B[kIdx * N + nIdx];
            }
#pragma unroll
            for (int i = 0; i < 16; ++i) {
                const int  kk = kk0 + 2 * i;
                const bool ok = (k0 + kk < K) && (gn < N);
                Bs[n][kk] = (_Float16)(ok ? v[i] : 0.0f);
            }
        } else {
            // B[N,K] row-major: float2 along k, packed f16 stores
            const int  gk  = k0 + c2;
            const bool kOK = gk < K;
            const long kIdx = kOK ? gk : 0;
            float2 v[8];
#pragma unroll
            for (int i = 0; i < 8; ++i) {
                const int  gn   = bn + r0 + 16 * i;
                const long nIdx = (gn < N) ? gn : (N - 1);
                v[i] = *(const float2*)(B + nIdx * K + kIdx);
            }
#pragma unroll
            for (int i = 0; i < 8; ++i) {
                const int  gn = bn + r0 + 16 * i;
                const bool ok = kOK && (gn < N);
                H2 p;
                p.h[0] = (_Float16)(ok ? v[i].x : 0.0f);
                p.h[1] = (_Float16)(ok ? v[i].y : 0.0f);
                *(unsigned int*)&Bs[r0 + 16 * i][c2] = p.u;
            }
        }
        __syncthreads();

        // ---- fragments from LDS ----
        // A 16x32 f16: lane(m=l16,g): K = {8g..8g+7} then {16+8g..16+8g+7}
        union { v16h h; uint4 u[2]; } fa, fb[4];
        const _Float16* arow = &As[wm * 16 + l16][0];
        fa.u[0] = *(const uint4*)(arow + 8 * g);
        fa.u[1] = *(const uint4*)(arow + 16 + 8 * g);
        // B 32x16 f16: lane(n=l16,g): K = {8g..8g+7} then {16+8g..16+8g+7}
#pragma unroll
        for (int j = 0; j < 4; ++j) {
            const _Float16* brow = &Bs[wn * 64 + 16 * j + l16][0];
            fb[j].u[0] = *(const uint4*)(brow + 8 * g);
            fb[j].u[1] = *(const uint4*)(brow + 16 + 8 * g);
        }
#pragma unroll
        for (int j = 0; j < 4; ++j) {
            acc[j] = __builtin_amdgcn_wmma_f32_16x16x32_f16(
                         false, fa.h, false, fb[j].h, (short)0, acc[j],
                         false, false);
        }
        __syncthreads();
    }

    // ---- store: C/D layout: lane(n=l16, mo=8g); VGPR r -> row mo+r ----
    const int mo   = g * 8;
    const int row0 = bm + wm * 16 + mo;
    const int colb = bn + wn * 64 + l16;
#pragma unroll
    for (int r = 0; r < 8; ++r) {
        const int row = row0 + r;
        if (row < M) {
#pragma unroll
            for (int j = 0; j < 4; ++j) {
                const int col = colb + 16 * j;
                if (col < N) {
                    float v = acc[j][r];
                    if (bias) v += bias[col];
                    C[(long)row * N + col] = v;
                }
            }
        }
    }
}

// ---------------------------------------------------------------------------
// Edge message scatter: agg[dst] += m_lin[src] * w   (128 feats per edge)
// One 32-lane slice per edge, float4 per lane, hardware f32 atomics into the
// L2-resident agg buffer (51 MB << 192 MB L2).
// ---------------------------------------------------------------------------
__global__ __launch_bounds__(256) void edge_scatter(
    const float* __restrict__ mlin, const int* __restrict__ src,
    const int* __restrict__ dst, const float* __restrict__ ew,
    float* __restrict__ agg, int nEdges)
{
    const long gid = (long)blockIdx.x * blockDim.x + threadIdx.x;
    const long e  = gid >> 5;
    const int  l  = (int)(gid & 31);
    if (e >= nEdges) return;
    const int   s = src[e];
    const int   d = dst[e];
    const float w = ew[e];
    const float4 v = *(const float4*)(mlin + (long)s * GNHID + l * 4);
    float* out = agg + (long)d * GNHID + l * 4;
    unsafeAtomicAdd(out + 0, v.x * w);
    unsafeAtomicAdd(out + 1, v.y * w);
    unsafeAtomicAdd(out + 2, v.z * w);
    unsafeAtomicAdd(out + 3, v.w * w);
}

// ---------------------------------------------------------------------------
// Elementwise kernels
// ---------------------------------------------------------------------------
__global__ void zerok(float* __restrict__ p, int n) {
    int i = blockIdx.x * blockDim.x + threadIdx.x;
    if (i < n) p[i] = 0.0f;
}
__global__ void copyk(float* __restrict__ d, const float* __restrict__ s, int n) {
    int i = blockIdx.x * blockDim.x + threadIdx.x;
    if (i < n) d[i] = s[i];
}
__global__ void addk(float* __restrict__ d, const float* __restrict__ s, int n) {
    int i = blockIdx.x * blockDim.x + threadIdx.x;
    if (i < n) d[i] += s[i];
}

// GRU blend: gi,gh are [N,384] pre-activations (r,z,n gate order), h is [N,128]
__global__ void gru_elem(const float* __restrict__ gi, const float* __restrict__ gh,
                         const float* __restrict__ h, float* __restrict__ out, int n)
{
    int i = blockIdx.x * blockDim.x + threadIdx.x;
    if (i >= n) return;
    const int  row = i >> 7;
    const int  col = i & 127;
    const long b   = (long)row * (3 * GNHID);
    const float ir = gi[b + col],            hr = gh[b + col];
    const float iz = gi[b + GNHID + col],    hz = gh[b + GNHID + col];
    const float in_= gi[b + 2*GNHID + col],  hn = gh[b + 2*GNHID + col];
    const float r = 1.0f / (1.0f + __expf(-(ir + hr)));
    const float z = 1.0f / (1.0f + __expf(-(iz + hz)));
    const float nn = tanhf(in_ + r * hn);
    out[i] = (1.0f - z) * nn + z * h[i];
}

// In-place log_softmax over rows of 40; one wave32 per row.
__global__ __launch_bounds__(256) void logsoftmax40(float* __restrict__ logits, int M)
{
    const int row  = blockIdx.x * 8 + (threadIdx.x >> 5);
    const int lane = threadIdx.x & 31;
    if (row >= M) return;
    float* p = logits + (long)row * GNCLASS;
    const float a = p[lane];                                  // lane 0..31 < 40
    const float b = (lane < 8) ? p[lane + 32] : -3.0e38f;
    float m = fmaxf(a, b);
#pragma unroll
    for (int o = 16; o > 0; o >>= 1) m = fmaxf(m, __shfl_xor(m, o, 32));
    float s = __expf(a - m) + ((lane < 8) ? __expf(b - m) : 0.0f);
#pragma unroll
    for (int o = 16; o > 0; o >>= 1) s += __shfl_xor(s, o, 32);
    const float lse = m + logf(s);
    p[lane] = a - lse;
    if (lane < 8) p[lane + 32] = b - lse;
}

// ---------------------------------------------------------------------------
// Host orchestration
// ---------------------------------------------------------------------------
static inline void launch_gemm(float* C, const float* A, const float* B,
                               const float* bias, int M, int N, int K, int opB,
                               hipStream_t stream)
{
    dim3 grid((M + BM - 1) / BM, (N + BN - 1) / BN);
    wmma_gemm_bias<<<grid, dim3(256), 0, stream>>>(A, B, bias, C, M, N, K, opB);
}

extern "C" void kernel_launch(void* const* d_in, const int* in_sizes, int n_in,
                              void* d_out, int out_size, void* d_ws, size_t ws_size,
                              hipStream_t stream)
{
    (void)in_sizes; (void)n_in; (void)out_size; (void)ws_size;

    const float* x       = (const float*)d_in[0];
    const int*   eidx    = (const int*)d_in[1];
    const float* ew      = (const float*)d_in[2];
    const float* embW    = (const float*)d_in[3];
    const float* embB    = (const float*)d_in[4];
    const float* convW   = (const float*)d_in[5];
    const float* convWih = (const float*)d_in[6];
    const float* convWhh = (const float*)d_in[7];
    const float* convBih = (const float*)d_in[8];
    const float* convBhh = (const float*)d_in[9];
    const float* rnnWih  = (const float*)d_in[10];
    const float* rnnWhh  = (const float*)d_in[11];
    const float* rnnBih  = (const float*)d_in[12];
    const float* rnnBhh  = (const float*)d_in[13];
    const float* mlpW    = (const float*)d_in[14];
    const float* mlpB    = (const float*)d_in[15];

    const int NN = GN_NODES, NE = GN_EDGES, NF = GNFEAT, NH = GNHID, NC = GNCLASS;
    const int* srcIdx = eidx;
    const int* dstIdx = eidx + NE;

    // workspace layout (fp32)
    float* ws  = (float*)d_ws;
    float* h   = ws;  ws += (long)NN * NH;       // hidden state
    float* hin = ws;  ws += (long)NN * NH;       // residual copy
    float* tmp = ws;  ws += (long)NN * NH;       // m_lin / h_conv
    float* agg = ws;  ws += (long)NN * NH;       // scatter target
    float* gi  = ws;  ws += (long)NN * 3 * NH;   // input-gate preact
    float* gh  = ws;  ws += (long)NN * 3 * NH;   // hidden-gate preact

    const int nElem  = NN * NH;                              // 12.8M
    const int ewBlks = (nElem + 255) / 256;
    const long edgeThreads = (long)NE * 32;
    const int edgeBlks = (int)((edgeThreads + 255) / 256);

    // 1) h = x @ emb_W + emb_b
    launch_gemm(h, x, embW, embB, NN, NH, NF, 0, stream);
    // 2) h_in = h
    copyk<<<ewBlks, 256, 0, stream>>>(hin, h, nElem);

    for (int i = 0; i < GNLAYERS; ++i) {
        const float* Wm   = convW   + (long)i * NH * NH;
        const float* Wih  = convWih + (long)i * 3 * NH * NH;
        const float* Whh  = convWhh + (long)i * 3 * NH * NH;
        const float* bih  = convBih + (long)i * 3 * NH;
        const float* bhh  = convBhh + (long)i * 3 * NH;

        // m_lin = h @ conv_W[i]
        launch_gemm(tmp, h, Wm, nullptr, NN, NH, NH, 0, stream);
        // agg = segment_sum(m_lin[src] * w, dst)
        zerok<<<ewBlks, 256, 0, stream>>>(agg, nElem);
        edge_scatter<<<edgeBlks, 256, 0, stream>>>(tmp, srcIdx, dstIdx, ew, agg, NE);

        // inner GRU: h_conv = GRU(agg, h)   (x @ W^T -> opB=1)
        launch_gemm(gi, agg, Wih, bih, NN, 3 * NH, NH, 1, stream);
        launch_gemm(gh, h,   Whh, bhh, NN, 3 * NH, NH, 1, stream);
        gru_elem<<<ewBlks, 256, 0, stream>>>(gi, gh, h, tmp, nElem);   // tmp = h_conv

        // outer GRU: h = GRU(h_conv, h)
        launch_gemm(gi, tmp, rnnWih, rnnBih, NN, 3 * NH, NH, 1, stream);
        launch_gemm(gh, h,   rnnWhh, rnnBhh, NN, 3 * NH, NH, 1, stream);
        gru_elem<<<ewBlks, 256, 0, stream>>>(gi, gh, h, h, nElem);     // in-place ok
    }

    // residual
    addk<<<ewBlks, 256, 0, stream>>>(h, hin, nElem);
    // logits = h @ mlp_W + mlp_b  -> d_out
    launch_gemm((float*)d_out, h, mlpW, mlpB, NN, NC, NH, 0, stream);
    // in-place log_softmax rows of 40
    logsoftmax40<<<(NN + 7) / 8, 256, 0, stream>>>((float*)d_out, NN);
}
